// AlibiBlock_532575944847
// MI455X (gfx1250) — matrix operand order repaired and checked by reading, not compiled, not run on hardware
//
#include <hip/hip_runtime.h>
#include <hip/hip_bf16.h>
#include <math.h>
#include <type_traits>

// ---------------------------------------------------------------------------
// Problem constants (reference: B=2, T=2048, C=1024, H=16, D=64)
// ---------------------------------------------------------------------------
#define BB 2
#define TT 2048
#define CC 1024
#define HH 16
#define DH 64
#define MM (BB * TT)          // 4096 rows

typedef __attribute__((ext_vector_type(16))) __bf16 v16bf;
typedef __attribute__((ext_vector_type(8)))  float  v8f;

union FragU { v16bf v; unsigned short u[16]; };

__device__ __forceinline__ unsigned short f32_to_bf16(float f) {
  unsigned int u = __builtin_bit_cast(unsigned int, f);
  u = u + 0x7FFFu + ((u >> 16) & 1u);   // round-to-nearest-even
  return (unsigned short)(u >> 16);
}

__device__ __forceinline__ v8f wmma_bf16(v16bf a, v16bf b, v8f c) {
  return __builtin_amdgcn_wmma_f32_16x16x32_bf16(false, a, false, b, (short)0, c,
                                                 false, false);
}

// Async copy global -> LDS (16 bytes per lane), tracked by ASYNCcnt.
__device__ __forceinline__ void async_copy_b128(unsigned lds_off,
                                                const void* base,
                                                unsigned byte_off) {
  asm volatile("global_load_async_to_lds_b128 %0, %1, %2"
               :: "v"(lds_off), "v"(byte_off), "s"(base) : "memory");
}
__device__ __forceinline__ void async_wait0() {
  asm volatile("s_wait_asynccnt 0x0" ::: "memory");
}
__device__ __forceinline__ void async_wait8() {   // allow 8 in flight
  asm volatile("s_wait_asynccnt 0x8" ::: "memory");
}

// A-matrix 16x32 bf16 fragment from row-major [m][k] storage (ISA 7.12.2):
//   lanes 0-15 = rows, VGPR i: k = (i>=4?16:0) + half*8 + 2*(i&3)
__device__ __forceinline__ v16bf frag_a(const unsigned short* p, int ld, int lane) {
  FragU f;
  const int r = lane & 15, hf = (lane >> 4) & 1;
  const unsigned short* row = p + r * ld;
#pragma unroll
  for (int i = 0; i < 8; ++i) {
    int k = ((i >> 2) << 4) + hf * 8 + ((i & 3) << 1);
    f.u[2 * i]     = row[k];
    f.u[2 * i + 1] = row[k + 1];
  }
  return f.v;
}

// B-matrix 32x16 bf16 fragment; lane holds a column of B. Source is B^T
// row-major: element B[k][n] at p[n*ld + k]  (ISA 7.12.4 B layout):
//   VGPR i: k = (i>=4?16:0) + half*8 + 2*(i&3)
__device__ __forceinline__ v16bf frag_b_t(const unsigned short* p, int ld, int lane) {
  FragU f;
  const int n = lane & 15, hf = (lane >> 4) & 1;
  const unsigned short* col = p + n * ld;
#pragma unroll
  for (int i = 0; i < 8; ++i) {
    int k = ((i >> 2) << 4) + hf * 8 + ((i & 3) << 1);
    f.u[2 * i]     = col[k];
    f.u[2 * i + 1] = col[k + 1];
  }
  return f.v;
}

// ---------------------------------------------------------------------------
// Weight transpose + f32 -> bf16 convert:  src[K][N] f32  ->  dst[N][K] bf16
// ---------------------------------------------------------------------------
__global__ __launch_bounds__(256) void transpose_bf16_kernel(
    const float* __restrict__ src, unsigned short* __restrict__ dst,
    int K, int N) {
  __shared__ float tile[32][33];
  const int tx = threadIdx.x & 31;
  const int ty = threadIdx.x >> 5;      // 0..7
  const int n0 = blockIdx.x * 32;
  const int k0 = blockIdx.y * 32;
#pragma unroll
  for (int i = 0; i < 4; ++i)
    tile[ty + i * 8][tx] = src[(size_t)(k0 + ty + i * 8) * N + n0 + tx];
  __syncthreads();
#pragma unroll
  for (int i = 0; i < 4; ++i)
    dst[(size_t)(n0 + ty + i * 8) * K + k0 + tx] = f32_to_bf16(tile[tx][ty + i * 8]);
}

// ---------------------------------------------------------------------------
// LayerNorm over C=1024, f32 in -> bf16 out
// ---------------------------------------------------------------------------
__global__ __launch_bounds__(256) void layernorm_bf16_kernel(
    const float* __restrict__ x, const float* __restrict__ w,
    const float* __restrict__ bias, unsigned short* __restrict__ out) {
  const int row  = blockIdx.x;
  const int tid  = threadIdx.x;
  const int lane = tid & 31;
  const int wid  = tid >> 5;
  const float* xr = x + (size_t)row * CC;
  const float4 xa = *(const float4*)(&xr[tid * 4]);
  float s  = xa.x + xa.y + xa.z + xa.w;
  float s2 = xa.x * xa.x + xa.y * xa.y + xa.z * xa.z + xa.w * xa.w;
#pragma unroll
  for (int d = 1; d < 32; d <<= 1) {
    s  += __shfl_xor(s,  d, 32);
    s2 += __shfl_xor(s2, d, 32);
  }
  __shared__ float ps[8], ps2[8];
  if (lane == 0) { ps[wid] = s; ps2[wid] = s2; }
  __syncthreads();
  float tot = 0.f, tot2 = 0.f;
#pragma unroll
  for (int i = 0; i < 8; ++i) { tot += ps[i]; tot2 += ps2[i]; }
  const float mean = tot * (1.0f / CC);
  const float var  = tot2 * (1.0f / CC) - mean * mean;
  const float rstd = rsqrtf(var + 1e-5f);
  const int i0 = tid * 4;
  const float4 wv = *(const float4*)(&w[i0]);
  const float4 bv = *(const float4*)(&bias[i0]);
  unsigned short* o = out + (size_t)row * CC + i0;
  o[0] = f32_to_bf16((xa.x - mean) * rstd * wv.x + bv.x);
  o[1] = f32_to_bf16((xa.y - mean) * rstd * wv.y + bv.y);
  o[2] = f32_to_bf16((xa.z - mean) * rstd * wv.z + bv.z);
  o[3] = f32_to_bf16((xa.w - mean) * rstd * wv.w + bv.w);
}

// ---------------------------------------------------------------------------
// Tiled WMMA GEMM:  out[M][N] = A[M][K](bf16) * Bt[N][K]^T(bf16) + bias  (+epi)
//   EPI 0: out bf16 = acc + bias
//   EPI 1: out f32  = resid + acc + bias
//   EPI 2: out bf16 = gelu(acc + bias)   (exact erf)
// Block tile 128x128, K-step 64, DOUBLE-BUFFERED async-to-LDS staging:
// stage s+1 is issued before computing stage s; `s_wait_asynccnt 8` retires
// only the 8 oldest (in-order) copies = the current stage, keeping the next
// stage in flight behind the WMMAs. 8 waves each compute 32x64.
// ---------------------------------------------------------------------------
template <int EPI>
__global__ __launch_bounds__(256) void gemm_bf16_kernel(
    const unsigned short* __restrict__ A,
    const unsigned short* __restrict__ Bt,
    const float* __restrict__ bias,
    const float* __restrict__ resid,
    void* __restrict__ outp,
    int M, int N, int K) {
  constexpr int BK  = 64;
  constexpr int LDT = 72;               // padded stride (bank-conflict-free)
  __shared__ __align__(16) unsigned short sA[2][128 * LDT];
  __shared__ __align__(16) unsigned short sB[2][128 * LDT];
  const int tid  = threadIdx.x;
  const int lane = tid & 31;
  const int wid  = tid >> 5;
  const int wm   = wid & 3;             // 4 waves along M
  const int wn   = wid >> 2;            // 2 waves along N
  const int m0   = blockIdx.y * 128;
  const int n0   = blockIdx.x * 128;

  v8f acc[2][4] = {};

  // issue one stage of async copies (8 ops/thread, 4 units per tile)
  auto stage = [&](int buf, int k0) {
#pragma unroll
    for (int u = 0; u < 4; ++u) {
      const int unit = tid + u * 256;
      const int r = unit >> 3;
      const int c = (unit & 7) << 3;
      const size_t oa = (m0 + r) * (size_t)K + k0 + c;
      const size_t ob = (n0 + r) * (size_t)K + k0 + c;
      async_copy_b128((unsigned)(size_t)&sA[buf][r * LDT + c], A,  (unsigned)(oa * 2u));
      async_copy_b128((unsigned)(size_t)&sB[buf][r * LDT + c], Bt, (unsigned)(ob * 2u));
      if (k0 + BK < K) {                // warm GL2 one further stage ahead
        __builtin_prefetch(&A[oa + BK], 0, 0);
        __builtin_prefetch(&Bt[ob + BK], 0, 0);
      }
    }
  };

  const int nstages = K / BK;
  stage(0, 0);
  for (int s = 0; s < nstages; ++s) {
    const int cur = s & 1;
    const bool more = (s + 1) < nstages;
    if (more) stage(cur ^ 1, (s + 1) * BK);   // overlap next stage with compute
    if (more) async_wait8(); else async_wait0();
    __syncthreads();
#pragma unroll
    for (int c2 = 0; c2 < 2; ++c2) {    // two K=32 chunks per stage
      v16bf af[2], bf[4];
#pragma unroll
      for (int t = 0; t < 2; ++t)
        af[t] = frag_a(&sA[cur][(wm * 32 + t * 16) * LDT + c2 * 32], LDT, lane);
#pragma unroll
      for (int t = 0; t < 4; ++t)
        bf[t] = frag_b_t(&sB[cur][(wn * 64 + t * 16) * LDT + c2 * 32], LDT, lane);
#pragma unroll
      for (int t = 0; t < 2; ++t)
#pragma unroll
        for (int v = 0; v < 4; ++v)
          acc[t][v] = wmma_bf16(af[t], bf[v], acc[t][v]);
    }
    __syncthreads();
  }

  const int hf = (lane >> 4) & 1;
  const int cn = lane & 15;
#pragma unroll
  for (int s = 0; s < 2; ++s)
#pragma unroll
    for (int t = 0; t < 4; ++t)
#pragma unroll
      for (int j = 0; j < 8; ++j) {
        const int m = m0 + wm * 32 + s * 16 + j + hf * 8;
        const int n = n0 + wn * 64 + t * 16 + cn;
        const size_t idx = (size_t)m * N + n;
        float v = acc[s][t][j] + bias[n];
        if constexpr (EPI == 0) {
          ((unsigned short*)outp)[idx] = f32_to_bf16(v);
        } else if constexpr (EPI == 1) {
          ((float*)outp)[idx] = resid[idx] + v;
        } else {
          const float g = 0.5f * v * (1.0f + erff(v * 0.70710678118654752f));
          ((unsigned short*)outp)[idx] = f32_to_bf16(g);
        }
      }
}

// ---------------------------------------------------------------------------
// Flash-style causal attention with ALiBi. One block = (q-tile of 64, b*h).
// 4 waves; each wave owns a 16x64 Q tile; online softmax in f32 accumulators.
// K staged via async-to-LDS; V staged TRANSPOSED (manual scatter).
// Causal mask evaluated only on the diagonal tile; ALiBi bias is hoisted to
// per-row / per-column linear terms (slope*q const, slope*k per tile).
// qkv: [B*T, 3C] bf16 (q | k | v)   ->   y: [B*T, C] bf16
// ---------------------------------------------------------------------------
__global__ __launch_bounds__(128) void attn_kernel(
    const unsigned short* __restrict__ qkv, unsigned short* __restrict__ y) {
  constexpr int LD  = 3 * CC;
  constexpr int LDT = 72;               // padded LDS stride
  __shared__ __align__(16) unsigned short sK [64 * LDT];   // [t][d]
  __shared__ __align__(16) unsigned short sVt[64 * LDT];   // [d][t] (transposed)
  __shared__ __align__(16) unsigned short sP [4][16 * LDT];

  const int tid  = threadIdx.x;
  const int lane = tid & 31;
  const int wid  = tid >> 5;                 // 0..3
  const int qt   = blockIdx.x;               // q tile (64 rows)
  const int bh   = blockIdx.y;
  const int b    = bh >> 4;
  const int h    = bh & 15;
  const float slope = exp2f(-0.5f * (float)(h + 1));  // 16 heads: 2^(-(i+1)/2)
  const int hf = (lane >> 4) & 1;
  const int cn = lane & 15;

  const size_t rowbase = (size_t)b * TT;
  const int q0 = qt * 64 + wid * 16;

  // Preload Q fragments (16x64 A-matrix -> two K=32 chunks)
  v16bf qf[2];
  {
    const unsigned short* qptr = qkv + (rowbase + q0) * LD + h * DH;
#pragma unroll
    for (int c = 0; c < 2; ++c) qf[c] = frag_a(qptr + c * 32, LD, lane);
  }

  v8f o[4] = {};
  float mst[8], lst[8], qb[8];
#pragma unroll
  for (int j = 0; j < 8; ++j) {
    mst[j] = -3.0e38f;
    lst[j] = 0.0f;
    qb[j]  = slope * (float)(q0 + j + hf * 8);   // ALiBi row term (constant)
  }

  auto process_tile = [&](int kt, auto MASKED) {
    // Cooperatively stage K (async) and V (transposed scatter) 64x64 tiles
#pragma unroll
    for (int u = 0; u < 4; ++u) {
      const int unit = tid + u * 128;        // 0..511
      const int r = unit >> 3;               // t within tile
      const int c = (unit & 7) << 3;         // d base
      const size_t g = (rowbase + kt * 64 + r) * LD + h * DH + c;
      async_copy_b128((unsigned)(size_t)&sK[r * LDT + c], qkv,
                      (unsigned)((g + CC) * 2u));
      uint4 vraw = *(const uint4*)(&qkv[g + 2 * CC]);
      const unsigned short* vp = (const unsigned short*)&vraw;
#pragma unroll
      for (int i = 0; i < 8; ++i) sVt[(c + i) * LDT + r] = vp[i];
      if (kt < qt) {
        __builtin_prefetch(&qkv[g + 64 * LD + CC], 0, 0);
        __builtin_prefetch(&qkv[g + 64 * LD + 2 * CC], 0, 0);
      }
    }
    async_wait0();
    __syncthreads();

    // S = Q K^T   (K tile rows are columns of S -> B^T layout, contiguous)
    v8f sacc[4] = {};
#pragma unroll
    for (int nt = 0; nt < 4; ++nt)
#pragma unroll
      for (int c = 0; c < 2; ++c) {
        const v16bf kf = frag_b_t(&sK[(nt * 16) * LDT + c * 32], LDT, lane);
        sacc[nt] = wmma_bf16(qf[c], kf, sacc[nt]);
      }

    // ALiBi column term for this tile (4 cvt+mul per tile, not per element)
    float kb[4];
#pragma unroll
    for (int nt = 0; nt < 4; ++nt)
      kb[nt] = slope * (float)(kt * 64 + nt * 16 + cn);

    float mnew[8], rsum[8];
#pragma unroll
    for (int j = 0; j < 8; ++j) mnew[j] = mst[j];
#pragma unroll
    for (int nt = 0; nt < 4; ++nt)
#pragma unroll
      for (int j = 0; j < 8; ++j) {
        float sv = fmaf(sacc[nt][j], 0.125f, qb[j] - kb[nt]);
        if constexpr (decltype(MASKED)::value) {
          const int qpos = q0 + j + hf * 8;
          const int kpos = kt * 64 + nt * 16 + cn;
          if (kpos > qpos) sv = -3.0e38f;    // causal mask (diagonal tile only)
        }
        sacc[nt][j] = sv;
        mnew[j] = fmaxf(mnew[j], sv);
      }
#pragma unroll
    for (int j = 0; j < 8; ++j)
#pragma unroll
      for (int d = 1; d < 16; d <<= 1)
        mnew[j] = fmaxf(mnew[j], __shfl_xor(mnew[j], d, 32));

#pragma unroll
    for (int j = 0; j < 8; ++j) rsum[j] = 0.0f;
#pragma unroll
    for (int nt = 0; nt < 4; ++nt)
#pragma unroll
      for (int j = 0; j < 8; ++j) {
        const float p = __expf(sacc[nt][j] - mnew[j]);
        rsum[j] += p;
        sP[wid][(j + hf * 8) * LDT + nt * 16 + cn] = f32_to_bf16(p);
      }
#pragma unroll
    for (int j = 0; j < 8; ++j) {
#pragma unroll
      for (int d = 1; d < 16; d <<= 1) rsum[j] += __shfl_xor(rsum[j], d, 32);
      const float scale = __expf(mst[j] - mnew[j]);
      lst[j] = lst[j] * scale + rsum[j];
      mst[j] = mnew[j];
#pragma unroll
      for (int dt = 0; dt < 4; ++dt) o[dt][j] *= scale;
    }

    // O += P * V   (P from per-wave LDS as A; V^T rows contiguous as B^T)
#pragma unroll
    for (int c = 0; c < 2; ++c) {
      const v16bf pf = frag_a(&sP[wid][c * 32], LDT, lane);
#pragma unroll
      for (int dt = 0; dt < 4; ++dt) {
        const v16bf vf = frag_b_t(&sVt[(dt * 16) * LDT + c * 32], LDT, lane);
        o[dt] = wmma_bf16(pf, vf, o[dt]);
      }
    }
    __syncthreads();
  };

  for (int kt = 0; kt < qt; ++kt) process_tile(kt, std::false_type{});
  process_tile(qt, std::true_type{});          // diagonal tile: masked

  // y = O / l
#pragma unroll
  for (int dt = 0; dt < 4; ++dt)
#pragma unroll
    for (int j = 0; j < 8; ++j) {
      const float val = o[dt][j] / lst[j];
      const size_t gr = rowbase + q0 + j + hf * 8;
      y[gr * CC + h * DH + dt * 16 + cn] = f32_to_bf16(val);
    }
}

// ---------------------------------------------------------------------------
// Host orchestration
// ---------------------------------------------------------------------------
extern "C" void kernel_launch(void* const* d_in, const int* in_sizes, int n_in,
                              void* d_out, int out_size, void* d_ws, size_t ws_size,
                              hipStream_t stream) {
  const float* x      = (const float*)d_in[0];
  const float* ln1_w  = (const float*)d_in[1];
  const float* ln1_b  = (const float*)d_in[2];
  const float* w_qkv  = (const float*)d_in[3];
  const float* b_qkv  = (const float*)d_in[4];
  const float* w_proj = (const float*)d_in[5];
  const float* b_proj = (const float*)d_in[6];
  const float* ln2_w  = (const float*)d_in[7];
  const float* ln2_b  = (const float*)d_in[8];
  const float* w_fc   = (const float*)d_in[9];
  const float* b_fc   = (const float*)d_in[10];
  const float* w_fc2  = (const float*)d_in[11];
  const float* b_fc2  = (const float*)d_in[12];
  float* out = (float*)d_out;

  char* ws = (char*)d_ws;
  // scratch layout (byte offsets)
  unsigned short* wqkvT  = (unsigned short*)(ws + 0);           // 3072x1024 bf16
  unsigned short* wprojT = (unsigned short*)(ws + 6291456);     // 1024x1024 bf16
  unsigned short* wfcT   = (unsigned short*)(ws + 8388608);     // 4096x1024 bf16
  unsigned short* wfc2T  = (unsigned short*)(ws + 16777216);    // 1024x4096 bf16
  unsigned short* hbuf   = (unsigned short*)(ws + 25165824);    // 4096x1024 bf16
  unsigned short* qkvb   = (unsigned short*)(ws + 33554432);    // 4096x3072 bf16
  unsigned short* ybuf   = (unsigned short*)(ws + 58720256);    // 4096x1024 bf16
  float*          x2     = (float*)(ws + 67108864);             // 4096x1024 f32
  unsigned short* h3     = (unsigned short*)(ws + 83886080);    // 4096x4096 bf16

  // 1) transpose+convert all weights to bf16 [N][K]
  transpose_bf16_kernel<<<dim3(3072 / 32, 1024 / 32), 256, 0, stream>>>(w_qkv,  wqkvT,  1024, 3072);
  transpose_bf16_kernel<<<dim3(1024 / 32, 1024 / 32), 256, 0, stream>>>(w_proj, wprojT, 1024, 1024);
  transpose_bf16_kernel<<<dim3(4096 / 32, 1024 / 32), 256, 0, stream>>>(w_fc,   wfcT,   1024, 4096);
  transpose_bf16_kernel<<<dim3(1024 / 32, 4096 / 32), 256, 0, stream>>>(w_fc2,  wfc2T,  4096, 1024);

  // 2) LN1
  layernorm_bf16_kernel<<<MM, 256, 0, stream>>>(x, ln1_w, ln1_b, hbuf);

  // 3) QKV GEMM: [4096,1024] x [1024,3072] -> bf16
  gemm_bf16_kernel<0><<<dim3(3072 / 128, MM / 128), 256, 0, stream>>>(
      hbuf, wqkvT, b_qkv, nullptr, qkvb, MM, 3072, 1024);

  // 4) attention (flash, causal + ALiBi)
  attn_kernel<<<dim3(TT / 64, BB * HH), 128, 0, stream>>>(qkvb, ybuf);

  // 5) proj GEMM + residual -> x2 (f32)
  gemm_bf16_kernel<1><<<dim3(1024 / 128, MM / 128), 256, 0, stream>>>(
      ybuf, wprojT, b_proj, x, x2, MM, 1024, 1024);

  // 6) LN2
  layernorm_bf16_kernel<<<MM, 256, 0, stream>>>(x2, ln2_w, ln2_b, hbuf);

  // 7) FC GEMM + GELU -> bf16
  gemm_bf16_kernel<2><<<dim3(4096 / 128, MM / 128), 256, 0, stream>>>(
      hbuf, wfcT, b_fc, nullptr, h3, MM, 4096, 1024);

  // 8) FC2 GEMM + residual -> d_out (f32)
  gemm_bf16_kernel<1><<<dim3(1024 / 128, MM / 128), 256, 0, stream>>>(
      h3, wfc2T, b_fc2, x2, out, MM, 1024, 4096);
}
